// GIN_pyg_43997644981011
// MI455X (gfx1250) — matrix operand order repaired and checked by reading, not compiled, not run on hardware
//
#include <hip/hip_runtime.h>
#include <hip/hip_bf16.h>

typedef __attribute__((ext_vector_type(16))) __bf16 v16bf;
typedef __attribute__((ext_vector_type(8)))  float  v8f;

#define CH 64
#define WPK 4096   // bf16 elements per packed 64x64 weight matrix

// ---------------- degree / norm precompute ----------------
__global__ void k_deg_init(float* deg, int n) {
    int i = blockIdx.x * blockDim.x + threadIdx.x;
    if (i < n) deg[i] = 1.0f;           // self-loop weight
}

__global__ void k_deg_acc(const int* __restrict__ dst, float* deg, int e) {
    int i = blockIdx.x * blockDim.x + threadIdx.x;
    if (i < e) atomicAdd(&deg[dst[i]], 1.0f);
}

__global__ void k_dinv(float* deg, int n) {
    int i = blockIdx.x * blockDim.x + threadIdx.x;
    if (i < n) deg[i] = rsqrtf(deg[i]); // in place: deg -> d^-1/2
}

__global__ void k_norm(const int* __restrict__ src, const int* __restrict__ dst,
                       const float* __restrict__ dinv, float* __restrict__ nrm, int e) {
    int i = blockIdx.x * blockDim.x + threadIdx.x;
    if (i < e) nrm[i] = dinv[src[i]] * dinv[dst[i]];
}

// ---------------- one-time weight pack: fp32 64x64 -> bf16 in B-register layout ----
// Packed layout: Wp[((ct*2 + kh)*32 + lane)*16 + i], so a wave's B fragment for
// (col-tile ct, K-half kh) is one contiguous 32B v16bf load per lane.
__global__ void k_packW6(const float* __restrict__ Wa, const float* __restrict__ Wb,
                         const float* __restrict__ Wc, const float* __restrict__ Wd,
                         const float* __restrict__ We_, const float* __restrict__ Wf_,
                         __bf16* __restrict__ Wp) {
    const float* Ws[6] = {Wa, Wb, Wc, Wd, We_, Wf_};
    const float* W = Ws[blockIdx.x];
    __bf16* base = Wp + (size_t)blockIdx.x * WPK;

    int t = threadIdx.x;                 // 256 threads: (ct, kh, lane)
    int ct = t >> 6, kh = (t >> 5) & 1, lane = t & 31;
    int n = lane & 15, half = lane >> 4;
    __bf16* out = base + ((ct * 2 + kh) * 32 + lane) * 16;
#pragma unroll
    for (int i = 0; i < 8; ++i) {
        int k = kh * 32 + half * 8 + ((2 * i) & 7) + (i >> 2) * 16;
        out[2*i]   = (__bf16)W[(k)     * CH + ct * 16 + n];
        out[2*i+1] = (__bf16)W[(k + 1) * CH + ct * 16 + n];
    }
}

// ---------------- WMMA GEMM: Y[n x 64] = X[n x 64] @ W[64 x 64] ----------------
// One wave computes a 16-row x 64-col tile via 4 col-tiles x 2 K-steps of
// v_wmma_f32_16x16x32_bf16. Tile index is wave-uniform -> EXEC all ones.
#define PK8(vec, off, A, B)                                                     \
    vec[(off)+0]=(__bf16)A.x; vec[(off)+1]=(__bf16)A.y;                         \
    vec[(off)+2]=(__bf16)A.z; vec[(off)+3]=(__bf16)A.w;                         \
    vec[(off)+4]=(__bf16)B.x; vec[(off)+5]=(__bf16)B.y;                         \
    vec[(off)+6]=(__bf16)B.z; vec[(off)+7]=(__bf16)B.w;

__global__ void k_gemm_wmma(const float* __restrict__ X, const __bf16* __restrict__ Wp,
                            float* __restrict__ Y, int nrows) {
    const int lane = threadIdx.x & 31;
    const int wave = threadIdx.x >> 5;
    const int tile = blockIdx.x * (blockDim.x >> 5) + wave;   // 16-row tile
    if (tile * 16 >= nrows) return;                            // wave-uniform exit

    const int m    = lane & 15;
    const int half = lane >> 4;                                // 0: lanes 0-15, 1: lanes 16-31
    const bool full = (tile * 16 + 16 <= nrows);               // wave-uniform
    int row = tile * 16 + m;
    int lrow = row < nrows ? row : nrows - 1;

    // A matrix (16x32 bf16, ISA 7.12.2): per lane, four contiguous 8-float chunks
    const float4* xv = (const float4*)(X + (long long)lrow * CH);
    float4 p0 = xv[half * 2 + 0];       // K  0..7  (half-adjusted)
    float4 p1 = xv[half * 2 + 1];
    float4 p2 = xv[half * 2 + 4];       // K 16..23
    float4 p3 = xv[half * 2 + 5];
    float4 q0 = xv[half * 2 + 8];       // K 32..39
    float4 q1 = xv[half * 2 + 9];
    float4 q2 = xv[half * 2 + 12];      // K 48..55
    float4 q3 = xv[half * 2 + 13];

    v16bf a0, a1;
    PK8(a0, 0, p0, p1); PK8(a0, 8, p2, p3);
    PK8(a1, 0, q0, q1); PK8(a1, 8, q2, q3);

    const v16bf* Bp = (const v16bf*)Wp;
    const int n = lane & 15;
#pragma unroll
    for (int ct = 0; ct < 4; ++ct) {
        v16bf b0 = Bp[(ct * 2 + 0) * 32 + lane];
        v16bf b1 = Bp[(ct * 2 + 1) * 32 + lane];
        v8f c = {};
        c = __builtin_amdgcn_wmma_f32_16x16x32_bf16(false, a0, false, b0, (short)0, c, false, false);
        c = __builtin_amdgcn_wmma_f32_16x16x32_bf16(false, a1, false, b1, (short)0, c, false, false);
        // C layout: VGPR r -> row (half*8 + r), col = lane&15 within col-tile
        float* yout = Y + (long long)(tile * 16 + half * 8) * CH + ct * 16 + n;
        if (full) {
#pragma unroll
            for (int r = 0; r < 8; ++r) yout[r * CH] = c[r];
        } else {
#pragma unroll
            for (int r = 0; r < 8; ++r)
                if (tile * 16 + half * 8 + r < nrows) yout[r * CH] = c[r];
        }
    }
}

// ---------------- edge scatter: AGG[dst] += norm * H[src] ----------------
// 16 threads per edge, float4 gathers, f32 global atomics (bandwidth-dominant phase)
__global__ void k_scatter(const int* __restrict__ src, const int* __restrict__ dst,
                          const float* __restrict__ nrm, const float* __restrict__ H,
                          float* __restrict__ AGG, int e) {
    long long idx = (long long)blockIdx.x * blockDim.x + threadIdx.x;
    if (idx >= (long long)e * 16) return;
    int ed = (int)(idx >> 4);
    int c4 = (int)(idx & 15) * 4;
    int s = src[ed], d = dst[ed];
    float w = nrm[ed];
    const float4 h = *(const float4*)(H + (long long)s * CH + c4);
    float* out = AGG + (long long)d * CH + c4;
    atomicAdd(out + 0, w * h.x);
    atomicAdd(out + 1, w * h.y);
    atomicAdd(out + 2, w * h.z);
    atomicAdd(out + 3, w * h.w);
}

// ---------------- elementwise combines ----------------
// H1 = relu(AGG + selfnorm*HW1 + b1) + relu(HWe + be)
__global__ void k_combine1(const float* __restrict__ AGG, const float* __restrict__ HW1,
                           const float* __restrict__ HWe, const float* __restrict__ dinv,
                           const float* __restrict__ b1, const float* __restrict__ be,
                           float* __restrict__ H1, int n) {
    int idx = blockIdx.x * blockDim.x + threadIdx.x;
    if (idx >= n * CH) return;
    int i = idx >> 6, c = idx & 63;
    float dv = dinv[i];
    float g = AGG[idx] + dv * dv * HW1[idx] + b1[c];
    float d = HWe[idx] + be[c];
    H1[idx] = fmaxf(g, 0.0f) + fmaxf(d, 0.0f);
}

// HT {=, +=} relu(AGG + selfnorm*HW2 + b2) + H1
__global__ void k_combine2(const float* __restrict__ AGG, const float* __restrict__ HW2,
                           const float* __restrict__ H1, const float* __restrict__ dinv,
                           const float* __restrict__ b2, float* __restrict__ HT,
                           int n, int accum) {
    int idx = blockIdx.x * blockDim.x + threadIdx.x;
    if (idx >= n * CH) return;
    int i = idx >> 6, c = idx & 63;
    float dv = dinv[i];
    float g = fmaxf(AGG[idx] + dv * dv * HW2[idx] + b2[c], 0.0f) + H1[idx];
    if (accum) HT[idx] += g; else HT[idx] = g;
}

// ---------------- pooling + head ----------------
__global__ void k_pool(const float* __restrict__ HT, const int* __restrict__ batch,
                       float* __restrict__ sums, float* __restrict__ cnts, int n) {
    int idx = blockIdx.x * blockDim.x + threadIdx.x;
    if (idx >= n * CH) return;
    int i = idx >> 6, c = idx & 63;
    int g = batch[i];
    atomicAdd(&sums[g * CH + c], HT[idx]);
    if (c == 0) atomicAdd(&cnts[g], 1.0f);
}

__global__ void k_final(const float* __restrict__ sums, const float* __restrict__ cnts,
                        const float* __restrict__ Wf, const float* __restrict__ bF,
                        float* __restrict__ out, int G) {
    int g = blockIdx.x * blockDim.x + threadIdx.x;
    if (g >= G) return;
    float cnt = fmaxf(cnts[g], 1.0f);
    float acc = 0.0f;
#pragma unroll
    for (int c = 0; c < CH; ++c) acc += sums[g * CH + c] * Wf[c];
    out[g] = acc / cnt + bF[0];
}

// ---------------- host orchestration ----------------
extern "C" void kernel_launch(void* const* d_in, const int* in_sizes, int n_in,
                              void* d_out, int out_size, void* d_ws, size_t ws_size,
                              hipStream_t stream) {
    const float* x   = (const float*)d_in[0];
    const float* xsc = (const float*)d_in[1];
    const float* W1  = (const float*)d_in[2];
    const float* b1  = (const float*)d_in[3];
    const float* W2  = (const float*)d_in[4];
    const float* b2  = (const float*)d_in[5];
    const float* We  = (const float*)d_in[6];
    const float* be  = (const float*)d_in[7];
    const float* W1s = (const float*)d_in[8];
    const float* b1s = (const float*)d_in[9];
    const float* W2s = (const float*)d_in[10];
    const float* b2s = (const float*)d_in[11];
    const float* Wes = (const float*)d_in[12];
    const float* bes = (const float*)d_in[13];
    const float* Wf  = (const float*)d_in[14];
    const float* bF  = (const float*)d_in[15];
    const int*   ei  = (const int*)d_in[16];
    const int*   bat = (const int*)d_in[17];

    const int N = in_sizes[0] / CH;
    const int E = in_sizes[16] / 2;
    const int G = out_size;                 // num_graphs (output is [G,1])

    const int* src = ei;
    const int* dst = ei + E;

    // workspace carve-up (all 16B-aligned)
    float* w = (float*)d_ws;
    auto take = [&](size_t nf) { float* p = w; w += (nf + 3) & ~(size_t)3; return p; };
    float* dinv = take(N);
    float* nrm  = take(E);
    float* HW   = take((size_t)N * CH);
    float* HWe  = take((size_t)N * CH);
    float* AGG  = take((size_t)N * CH);
    float* H1   = take((size_t)N * CH);
    float* HT   = take((size_t)N * CH);
    float* sums = take((size_t)G * CH + G);   // cnts appended after sums
    float* cnts = sums + (size_t)G * CH;
    __bf16* Wpk = (__bf16*)take(6 * WPK / 2); // 6 packed 64x64 bf16 weight mats

    __bf16* Wp1  = Wpk + 0 * WPK;
    __bf16* Wp2  = Wpk + 1 * WPK;
    __bf16* Wpe  = Wpk + 2 * WPK;
    __bf16* Wp1s = Wpk + 3 * WPK;
    __bf16* Wp2s = Wpk + 4 * WPK;
    __bf16* Wpes = Wpk + 5 * WPK;

    const int TPB = 256;
    const int nodeBlocks = (N + TPB - 1) / TPB;
    const int edgeBlocks = (E + TPB - 1) / TPB;
    const int featBlocks = (N * CH + TPB - 1) / TPB;
    const long long scatThreads = (long long)E * 16;
    const int scatBlocks = (int)((scatThreads + TPB - 1) / TPB);
    const int tiles = (N + 15) / 16;
    const int gemmBlocks = (tiles + 7) / 8;     // 8 waves per 256-thread block

    // ---- one-time weight pack (bf16, B-register layout) ----
    k_packW6<<<6, 256, 0, stream>>>(W1, W2, We, W1s, W2s, Wes, Wpk);

    // ---- normalization precompute ----
    k_deg_init<<<nodeBlocks, TPB, 0, stream>>>(dinv, N);
    k_deg_acc <<<edgeBlocks, TPB, 0, stream>>>(dst, dinv, E);
    k_dinv    <<<nodeBlocks, TPB, 0, stream>>>(dinv, N);
    k_norm    <<<edgeBlocks, TPB, 0, stream>>>(src, dst, dinv, nrm, E);

    const size_t featBytes = (size_t)N * CH * sizeof(float);

    auto tower = [&](const float* X, const __bf16* Wa, const float* ba,
                     const __bf16* Wb, const float* bb,
                     const __bf16* Wc, const float* bc, int accum) {
        k_gemm_wmma<<<gemmBlocks, TPB, 0, stream>>>(X, Wa, HW, N);     // X @ W1
        k_gemm_wmma<<<gemmBlocks, TPB, 0, stream>>>(X, Wc, HWe, N);    // X @ We
        hipMemsetAsync(AGG, 0, featBytes, stream);
        k_scatter<<<scatBlocks, TPB, 0, stream>>>(src, dst, nrm, HW, AGG, E);
        k_combine1<<<featBlocks, TPB, 0, stream>>>(AGG, HW, HWe, dinv, ba, bc, H1, N);
        k_gemm_wmma<<<gemmBlocks, TPB, 0, stream>>>(H1, Wb, HW, N);    // H1 @ W2
        hipMemsetAsync(AGG, 0, featBytes, stream);
        k_scatter<<<scatBlocks, TPB, 0, stream>>>(src, dst, nrm, HW, AGG, E);
        k_combine2<<<featBlocks, TPB, 0, stream>>>(AGG, HW, H1, dinv, bb, HT, N, accum);
    };

    tower(x,   Wp1,  b1,  Wp2,  b2,  Wpe,  be,  0);
    tower(xsc, Wp1s, b1s, Wp2s, b2s, Wpes, bes, 1);

    // ---- mean pool per graph + final linear ----
    hipMemsetAsync(sums, 0, ((size_t)G * CH + G) * sizeof(float), stream);
    k_pool<<<featBlocks, TPB, 0, stream>>>(HT, bat, sums, cnts, N);
    k_final<<<(G + 63) / 64, 64, 0, stream>>>(sums, cnts, Wf, bF, (float*)d_out, G);
}
